// CrossAttentionFusion_2508260901449
// MI455X (gfx1250) — compile-verified, matrix operand order checked by reading, not run on hardware
//
#include <hip/hip_runtime.h>
#include <hip/hip_bf16.h>

// ---------------------------------------------------------------------------
// Fully-fused segmented cross-attention for MI455X (gfx1250).
// One workgroup per segment; whole segment's K (row-major) and V (transposed)
// live in LDS as bf16; flash attention with v_wmma_f32_16x16x32_bf16.
// Weights pre-converted to bf16 in workspace (L2-resident) by a tiny pre-pass.
// Next-tile feature rows are prefetched (global_prefetch_b8) under compute.
// ---------------------------------------------------------------------------

#define CDIM     128
#define LMAXSEG  512
#define TILE_M   128
#define THREADS  256
#define NWAVE    8

// LDS layout. All row strides are multiples of 8 bf16 elems (16 B) so operand
// fragments load as two ds_load_b128; stride mod 64 DWORDs == 4 so a 16-lane
// group's 4-DWORD accesses tile all 64 banks.
#define KS_STRIDE 136                        // bf16 elems per K row
#define VT_STRIDE 520                        // bf16 elems per Vt row
#define XQ_STRIDE 136                        // bf16 elems per staging row
#define P_STRIDE  40                         // bf16 elems per P row
#define KS_OFF    0
#define KS_BYTES  (LMAXSEG * KS_STRIDE * 2)               // 139264
#define VT_OFF    (KS_OFF + KS_BYTES)
#define VT_BYTES  (CDIM * VT_STRIDE * 2)                  // 133120
#define XQ_OFF    (VT_OFF + VT_BYTES)                     // 272384
#define XQ_BYTES  (TILE_M * XQ_STRIDE * 2)                // 34816
#define P_OFF     (XQ_OFF + XQ_BYTES)                     // 307200
#define P_WAVE_BYTES (16 * P_STRIDE * 2)                  // 1280
#define SMEM_BYTES (P_OFF + NWAVE * P_WAVE_BYTES)         // 317440

typedef __attribute__((ext_vector_type(16))) __bf16 v16bf;
typedef __attribute__((ext_vector_type(8)))  float  v8f;
typedef __attribute__((ext_vector_type(4)))  float  f32x4;   // clang vector
typedef __attribute__((ext_vector_type(4)))  unsigned int u32x4;

union FragAB {
  v16bf v;
  u32x4 q[2];
  unsigned int u[8];
};

// --- conversions ---
__device__ __forceinline__ unsigned short f2bf_rne(float f) {   // exact RNE
  unsigned int u = __float_as_uint(f);
  return (unsigned short)((u + 0x7FFFu + ((u >> 16) & 1u)) >> 16);
}
__device__ __forceinline__ unsigned short f2bf_rn(float f) {    // cheap RN
  return (unsigned short)((__float_as_uint(f) + 0x8000u) >> 16);
}
// Pack two f32 -> two bf16 (round-nearest) with one v_perm_b32.
__device__ __forceinline__ unsigned int pack2bf(float lo, float hi) {
  unsigned int a = __float_as_uint(lo) + 0x8000u;
  unsigned int b = __float_as_uint(hi) + 0x8000u;
  return __builtin_amdgcn_perm(b, a, 0x07060302u);
}

// 16x32 bf16 operand fragment from an LDS matrix stored [row][k] with 16B-
// aligned row stride: exactly two ds_load_b128 per lane.
// Lane l holds row (row0 + l&15); half = l>>4 selects the k sub-pattern.
__device__ __forceinline__ FragAB frag_from_lds(const unsigned short* base,
                                                int row0, int stride,
                                                int k0, int lane) {
  FragAB f;
  const unsigned short* rp =
      base + (row0 + (lane & 15)) * stride + k0 + ((lane >> 4) << 3);
  f.q[0] = *(const u32x4*)rp;
  f.q[1] = *(const u32x4*)(rp + 16);
  return f;
}

// 32x16 B fragment (Y = X*W^T) from a pre-converted bf16 weight matrix in
// global memory, stored [n][k] with stride CDIM: two global_load_b128.
__device__ __forceinline__ FragAB frag_from_Wbf(
    const unsigned short* __restrict__ W, int n0, int k0, int lane) {
  FragAB f;
  const unsigned short* rp =
      W + (n0 + (lane & 15)) * CDIM + k0 + ((lane >> 4) << 3);
  f.q[0] = *(const u32x4*)rp;
  f.q[1] = *(const u32x4*)(rp + 16);
  return f;
}

__device__ __forceinline__ v8f wmma_bf16(const FragAB& a, const FragAB& b, v8f c) {
  return __builtin_amdgcn_wmma_f32_16x16x32_bf16(false, a.v, false, b.v,
                                                 (short)0, c, false, false);
}

// Prefetch one 128-row f32 feature tile (512 B/row = 4 x 128B lines) into the
// cache hierarchy: emits global_prefetch_b8 (no counters, no VGPR return).
__device__ __forceinline__ void prefetch_tile(const float* __restrict__ F,
                                              long start, int rbase,
                                              int Ntot, int tid) {
  for (int e = tid; e < TILE_M * 4; e += THREADS) {
    const int r = e >> 2;
    const int c = (e & 3) << 5;            // 0,32,64,96 floats = 128B lines
    long grow = start + rbase + r;
    if (grow >= (long)Ntot) grow = (long)Ntot - 1;
    __builtin_prefetch(F + grow * CDIM + c, 0, 0);
  }
}

// ---------------------------------------------------------------------------
// Pre-pass: convert Wq|Wk|Wv (each 128x128 f32) to bf16 into workspace.
// ---------------------------------------------------------------------------
__global__ void cvt_weights(const float* __restrict__ Wq,
                            const float* __restrict__ Wk,
                            const float* __restrict__ Wv,
                            unsigned short* __restrict__ ws) {
  const int n = CDIM * CDIM;
  const float* src[3] = {Wq, Wk, Wv};
  for (int m = 0; m < 3; ++m) {
    const float* s = src[m];
    unsigned short* d = ws + m * n;
    for (int i = blockIdx.x * blockDim.x + threadIdx.x; i < n;
         i += gridDim.x * blockDim.x) {
      d[i] = f2bf_rne(s[i]);
    }
  }
}

// ---------------------------------------------------------------------------
// Main fused kernel.
// ---------------------------------------------------------------------------
__global__ __launch_bounds__(THREADS, 1)
void fused_seg_attn(const float* __restrict__ Qf, const float* __restrict__ Kf,
                    const unsigned short* __restrict__ Wbf,   // [3][128][128]
                    const float* __restrict__ bq,
                    const float* __restrict__ bk,
                    const float* __restrict__ bv,
                    const int* __restrict__ offset,
                    float* __restrict__ out, int Ntot) {
  extern __shared__ unsigned char smem[];
  unsigned short* Ks = (unsigned short*)(smem + KS_OFF);   // [token][ch]
  unsigned short* Vt = (unsigned short*)(smem + VT_OFF);   // [ch][token]
  unsigned short* XQ = (unsigned short*)(smem + XQ_OFF);   // staging / Q tile

  const unsigned short* WqB = Wbf;
  const unsigned short* WkB = Wbf + CDIM * CDIM;
  const unsigned short* WvB = Wbf + 2 * CDIM * CDIM;

  const int seg  = blockIdx.x;
  const int tid  = threadIdx.x;
  const int lane = tid & 31;
  const int wave = tid >> 5;
  const int half = lane >> 4;
  const int lm   = lane & 15;

  const long start = (seg == 0) ? 0L : (long)offset[seg - 1];
  const long endo  = (long)offset[seg];
  const int  len   = (int)(endo - start);
  const float scale = 0.088388347648318447f;   // 1/sqrt(128)

  unsigned short* Pw = (unsigned short*)(smem + P_OFF) + wave * 16 * P_STRIDE;

  // ============ Phase 1: project K and V for the whole segment ============
#pragma unroll 1
  for (int tile = 0; tile < LMAXSEG / TILE_M; ++tile) {
    const int rbase = tile * TILE_M;
    // Stage K_feature tile -> XQ as bf16 (non-temporal streaming read).
    for (int e = tid; e < TILE_M * (CDIM / 4); e += THREADS) {
      const int r  = e >> 5;
      const int c4 = (e & 31) << 2;
      long grow = start + rbase + r;            // reference clips to [0, N-1]
      if (grow >= (long)Ntot) grow = (long)Ntot - 1;
      const f32x4 x =
          __builtin_nontemporal_load((const f32x4*)(Kf + grow * CDIM + c4));
      uint2 p;
      p.x = pack2bf(x.x, x.y);
      p.y = pack2bf(x.z, x.w);
      *(uint2*)(XQ + r * XQ_STRIDE + c4) = p;
    }
    __syncthreads();

    // Pull the next K_feature tile toward the caches under the GEMM below.
    if (tile + 1 < LMAXSEG / TILE_M)
      prefetch_tile(Kf, start, rbase + TILE_M, Ntot, tid);

    const int srow = wave * 16;
    FragAB afr[4];
#pragma unroll
    for (int ks = 0; ks < 4; ++ks)
      afr[ks] = frag_from_lds(XQ, srow, XQ_STRIDE, ks * 32, lane);

    // K projection -> Ks (row-major)
#pragma unroll 1
    for (int cb = 0; cb < 8; ++cb) {
      v8f acc = {};
#pragma unroll
      for (int ks = 0; ks < 4; ++ks)
        acc = wmma_bf16(afr[ks], frag_from_Wbf(WkB, cb * 16, ks * 32, lane), acc);
      const float bias = bk[cb * 16 + lm];
      const int col = cb * 16 + lm;
#pragma unroll
      for (int r = 0; r < 8; ++r) {
        const int row = rbase + srow + r + half * 8;
        Ks[row * KS_STRIDE + col] = f2bf_rn(acc[r] + bias);
      }
    }
    // V projection -> Vt (transposed)
#pragma unroll 1
    for (int cb = 0; cb < 8; ++cb) {
      v8f acc = {};
#pragma unroll
      for (int ks = 0; ks < 4; ++ks)
        acc = wmma_bf16(afr[ks], frag_from_Wbf(WvB, cb * 16, ks * 32, lane), acc);
      const float bias = bv[cb * 16 + lm];
      const int col = cb * 16 + lm;
#pragma unroll
      for (int r = 0; r < 8; ++r) {
        const int row = rbase + srow + r + half * 8;
        Vt[col * VT_STRIDE + row] = f2bf_rn(acc[r] + bias);
      }
    }
    __syncthreads();
  }

  // First Q tile prefetch rides under the tail of phase 1 / first staging.
  prefetch_tile(Qf, start, 0, Ntot, tid);

  // ============ Phase 2: per Q tile -> project Q, flash attention ============
#pragma unroll 1
  for (int qt = 0; qt < LMAXSEG / TILE_M; ++qt) {
    const int qbase = qt * TILE_M;
    for (int e = tid; e < TILE_M * (CDIM / 4); e += THREADS) {
      const int r  = e >> 5;
      const int c4 = (e & 31) << 2;
      long grow = start + qbase + r;
      if (grow >= (long)Ntot) grow = (long)Ntot - 1;
      const f32x4 x =
          __builtin_nontemporal_load((const f32x4*)(Qf + grow * CDIM + c4));
      uint2 p;
      p.x = pack2bf(x.x, x.y);
      p.y = pack2bf(x.z, x.w);
      *(uint2*)(XQ + r * XQ_STRIDE + c4) = p;
    }
    __syncthreads();

    // Next Q tile streams in under the projection + attention below.
    if (qt + 1 < LMAXSEG / TILE_M)
      prefetch_tile(Qf, start, qbase + TILE_M, Ntot, tid);

    const int srow = wave * 16;
    {
      // Q projection in-place: preload all A-fragments of own strip (consumes
      // the full 16x128 strip) before overwriting with scaled bf16 Q.
      FragAB afr[4];
#pragma unroll
      for (int ks = 0; ks < 4; ++ks)
        afr[ks] = frag_from_lds(XQ, srow, XQ_STRIDE, ks * 32, lane);
#pragma unroll 1
      for (int cb = 0; cb < 8; ++cb) {
        v8f acc = {};
#pragma unroll
        for (int ks = 0; ks < 4; ++ks)
          acc = wmma_bf16(afr[ks], frag_from_Wbf(WqB, cb * 16, ks * 32, lane), acc);
        const float bias = bq[cb * 16 + lm];
        const int col = cb * 16 + lm;
#pragma unroll
        for (int r = 0; r < 8; ++r) {
          const int row = srow + r + half * 8;
          XQ[row * XQ_STRIDE + col] = f2bf_rn((acc[r] + bias) * scale);
        }
      }
    }
    __syncthreads();

    // ---- flash attention over this wave's 16-row Q strip ----
    FragAB qfr[4];
#pragma unroll
    for (int ks = 0; ks < 4; ++ks)
      qfr[ks] = frag_from_lds(XQ, srow, XQ_STRIDE, ks * 32, lane);

    v8f Oacc[8];
#pragma unroll
    for (int cb = 0; cb < 8; ++cb) Oacc[cb] = (v8f){};
    float rmax[8], rsum[8];
#pragma unroll
    for (int r = 0; r < 8; ++r) { rmax[r] = -3.0e38f; rsum[r] = 0.0f; }

#pragma unroll 1
    for (int kb = 0; kb < LMAXSEG / 32; ++kb) {
      const int kbase = kb * 32;
      if (kbase >= len) break;               // uniform ragged early-exit

      v8f s0 = {}, s1 = {};
#pragma unroll
      for (int ks = 0; ks < 4; ++ks) {
        FragAB b0 = frag_from_lds(Ks, kbase,      KS_STRIDE, ks * 32, lane);
        FragAB b1 = frag_from_lds(Ks, kbase + 16, KS_STRIDE, ks * 32, lane);
        s0 = wmma_bf16(qfr[ks], b0, s0);
        s1 = wmma_bf16(qfr[ks], b1, s1);
      }

      const int key0 = kbase + lm;
      const int key1 = key0 + 16;
#pragma unroll
      for (int r = 0; r < 8; ++r) {
        s0[r] = (key0 < len) ? s0[r] : -1.0e30f;
        s1[r] = (key1 < len) ? s1[r] : -1.0e30f;
      }

      // Online softmax; xor-shuffles stay within each 16-lane half, matching
      // the C/D striping (VGPR r: rows r / r+8, col = lane&15).
      float nm[8], fac[8];
#pragma unroll
      for (int r = 0; r < 8; ++r) {
        float t = fmaxf(s0[r], s1[r]);
        t = fmaxf(t, __shfl_xor(t, 1, 32));
        t = fmaxf(t, __shfl_xor(t, 2, 32));
        t = fmaxf(t, __shfl_xor(t, 4, 32));
        t = fmaxf(t, __shfl_xor(t, 8, 32));
        nm[r]  = fmaxf(rmax[r], t);
        fac[r] = __expf(rmax[r] - nm[r]);
        rmax[r] = nm[r];
      }
#pragma unroll
      for (int r = 0; r < 8; ++r) {
        const float p0 = __expf(s0[r] - nm[r]);
        const float p1 = __expf(s1[r] - nm[r]);
        s0[r] = p0; s1[r] = p1;
        float t = p0 + p1;
        t += __shfl_xor(t, 1, 32);
        t += __shfl_xor(t, 2, 32);
        t += __shfl_xor(t, 4, 32);
        t += __shfl_xor(t, 8, 32);
        rsum[r] = rsum[r] * fac[r] + t;
      }
#pragma unroll
      for (int cb = 0; cb < 8; ++cb)
#pragma unroll
        for (int r = 0; r < 8; ++r) Oacc[cb][r] *= fac[r];

      // Re-stripe P (C/D layout) -> A-fragment layout via per-wave LDS.
#pragma unroll
      for (int r = 0; r < 8; ++r) {
        const int prow = r + half * 8;
        Pw[prow * P_STRIDE + lm]      = f2bf_rn(s0[r]);
        Pw[prow * P_STRIDE + 16 + lm] = f2bf_rn(s1[r]);
      }
      FragAB pf = frag_from_lds(Pw, 0, P_STRIDE, 0, lane);

      // O += P @ V (contraction over this block's 32 keys).
#pragma unroll
      for (int cb = 0; cb < 8; ++cb) {
        FragAB bv_ = frag_from_lds(Vt, cb * 16, VT_STRIDE, kbase, lane);
        Oacc[cb] = wmma_bf16(pf, bv_, Oacc[cb]);
      }
    }

    // Epilogue: normalize and stream out (f32, non-temporal).
#pragma unroll 1
    for (int cb = 0; cb < 8; ++cb) {
      const int col = cb * 16 + lm;
#pragma unroll
      for (int r = 0; r < 8; ++r) {
        const int row = qbase + srow + r + half * 8;
        if (row < len) {
          const float inv = (rsum[r] > 0.0f) ? (1.0f / rsum[r]) : 0.0f;
          __builtin_nontemporal_store(Oacc[cb][r] * inv,
                                      out + (start + row) * CDIM + col);
        }
      }
    }
    __syncthreads();   // XQ reused by next Q tile
  }
}

extern "C" void kernel_launch(void* const* d_in, const int* in_sizes, int n_in,
                              void* d_out, int out_size, void* d_ws, size_t ws_size,
                              hipStream_t stream) {
  const float* Qf = (const float*)d_in[0];
  const float* Kf = (const float*)d_in[1];
  const float* Wq = (const float*)d_in[2];
  const float* bq = (const float*)d_in[3];
  const float* Wk = (const float*)d_in[4];
  const float* bk = (const float*)d_in[5];
  const float* Wv = (const float*)d_in[6];
  const float* bv = (const float*)d_in[7];
  const int* offset = (const int*)d_in[8];
  float* out = (float*)d_out;

  const int Ntot = in_sizes[0] / CDIM;
  const int B    = in_sizes[8];
  (void)n_in; (void)out_size; (void)ws_size;

  unsigned short* Wbf = (unsigned short*)d_ws;   // 3*128*128*2 = 98304 B

  cvt_weights<<<dim3(48), dim3(256), 0, stream>>>(Wq, Wk, Wv, Wbf);

  (void)hipFuncSetAttribute((const void*)fused_seg_attn,
                            hipFuncAttributeMaxDynamicSharedMemorySize,
                            (int)SMEM_BYTES);
  fused_seg_attn<<<dim3(B), dim3(THREADS), SMEM_BYTES, stream>>>(
      Qf, Kf, Wbf, bq, bk, bv, offset, out, Ntot);
}
